// GraphSAGEModel_1030792151555
// MI455X (gfx1250) — matrix operand order repaired and checked by reading, not compiled, not run on hardware
//
#include <hip/hip_runtime.h>
#include <hip/hip_bf16.h>

// ---------------- problem constants (match reference) ----------------
#define N_NODES 100000
#define DEGREE  16
#define FEAT    128
#define HID     64
#define OUTD    16
#define BATCH   4096
#define KNB     12

typedef __attribute__((ext_vector_type(16))) _Float16 v16h;
typedef __attribute__((ext_vector_type(8)))  _Float16 v8h;
typedef __attribute__((ext_vector_type(8)))  float    v8f;

// ---------------------------------------------------------------------
// Kernel 1: neighbor sampling. One thread per (b,k): computes idx1[b,k]
// and the 12 second-hop neighbors idx2[(b*K+k)*K + j].
// ---------------------------------------------------------------------
__global__ void sample_kernel(const int* __restrict__ idx0,
                              const int* __restrict__ buf1,
                              const int* __restrict__ buf2,
                              const int* __restrict__ indptr,
                              const int* __restrict__ indices,
                              const int* __restrict__ degrees,
                              int* __restrict__ idx1,
                              int* __restrict__ idx2) {
    int i = blockIdx.x * blockDim.x + threadIdx.x;   // i in [0, BATCH*KNB)
    if (i >= BATCH * KNB) return;
    int b = i / KNB;

    int n0   = idx0[b];
    int d0   = degrees[n0];
    int nb   = buf1[i] % d0;
    int n1   = indices[indptr[n0] + nb];
    idx1[i]  = n1;

    int d1   = degrees[n1];
    int base = indptr[n1];
    const int* brow = buf2 + i * KNB;
    int* orow = idx2 + i * KNB;
#pragma unroll
    for (int j = 0; j < KNB; ++j) {
        orow[j] = indices[base + (brow[j] % d1)];
    }
}

// ---------------------------------------------------------------------
// Kernel 2: convert enc_w [64x128] and h1_w [64x128] f32 -> f16
// ---------------------------------------------------------------------
__global__ void cvt_weights_kernel(const float* __restrict__ enc_w,
                                   const float* __restrict__ h1_w,
                                   _Float16* __restrict__ encw_h,
                                   _Float16* __restrict__ h1w_h) {
    int i = blockIdx.x * blockDim.x + threadIdx.x;
    if (i < HID * FEAT) {
        encw_h[i] = (_Float16)enc_w[i];
    } else if (i < 2 * HID * FEAT) {
        int j = i - HID * FEAT;
        h1w_h[j] = (_Float16)h1_w[j];
    }
}

// ---------------------------------------------------------------------
// Cooperative LDS fill: copy the 16 KB f16 weight tile into LDS once
// per workgroup (1024 uint4 vectors, 4 per thread at blockDim 256).
// ---------------------------------------------------------------------
__device__ __forceinline__ void fill_lds_weights(_Float16* sB,
                                                 const _Float16* __restrict__ w) {
    const uint4* src = (const uint4*)w;
    uint4*       dst = (uint4*)sB;
    const int nvec = HID * FEAT * (int)sizeof(_Float16) / 16;  // 1024
    for (int v = threadIdx.x; v < nvec; v += blockDim.x) dst[v] = src[v];
}

// ---------------------------------------------------------------------
// Kernel 3: Z = relu(emb @ enc_w^T)   [100000 x 64], f16 output.
// One wave per 16-row M-tile. N=64 -> 4 accumulator tiles.
// K=128 -> 4 x v_wmma_f32_16x16x32_f16 steps per N tile.
// Weights staged in LDS (DS pipe) so VMEM carries only the A-stream.
//
// A-fragment layout (16-bit A, 16x32): lanes 0-15 hold row M=lane with
// K = kk*32+0..15; lanes 16-31 hold same rows with
// K = kk*32+16..31  => per-lane k-base = kk*32 + (lane>>4)*16.
// B-fragment layout (32x16): lane L<16 holds column N=L, K=kk*32+0..15;
// lanes 16-31 hold K=kk*32+16..31 => 16 contiguous f16 from weight row N.
// ---------------------------------------------------------------------
__global__ void encode_kernel(const float* __restrict__ emb,
                              const _Float16* __restrict__ encw_h,
                              _Float16* __restrict__ Z,
                              int n_tiles) {
    __shared__ _Float16 sB[HID * FEAT];          // 16 KB
    fill_lds_weights(sB, encw_h);
    __syncthreads();

    int wave = threadIdx.x >> 5;
    int lane = threadIdx.x & 31;
    int tile = blockIdx.x * (blockDim.x >> 5) + wave;
    if (tile < n_tiles) {                        // wave-uniform guard, after barrier
        int mrow  = tile * 16 + (lane & 15);
        int khalf = lane >> 4;                   // 0 or 1
        int ncol  = lane & 15;

        v8f acc[4];
#pragma unroll
        for (int n = 0; n < 4; ++n) acc[n] = (v8f){};

#pragma unroll
        for (int kk = 0; kk < 4; ++kk) {
            int k0 = kk * 32 + khalf * 16;
            const float4* pa = (const float4*)(emb + (long)mrow * FEAT + k0);
            float4 c0 = pa[0];
            float4 c1 = pa[1];
            const float4* pb = (const float4*)(emb + (long)mrow * FEAT + k0 + 8);
            float4 c2 = pb[0];
            float4 c3 = pb[1];

            v16h a;
            a[0]  = (_Float16)c0.x;  a[1]  = (_Float16)c0.y;
            a[2]  = (_Float16)c0.z;  a[3]  = (_Float16)c0.w;
            a[4]  = (_Float16)c1.x;  a[5]  = (_Float16)c1.y;
            a[6]  = (_Float16)c1.z;  a[7]  = (_Float16)c1.w;
            a[8]  = (_Float16)c2.x;  a[9]  = (_Float16)c2.y;
            a[10] = (_Float16)c2.z;  a[11] = (_Float16)c2.w;
            a[12] = (_Float16)c3.x;  a[13] = (_Float16)c3.y;
            a[14] = (_Float16)c3.z;  a[15] = (_Float16)c3.w;

            int bk = kk * 32 + khalf * 16;
#pragma unroll
            for (int n = 0; n < 4; ++n) {
                v16h bfr = *(const v16h*)(sB + (n * 16 + ncol) * FEAT + bk);
                acc[n] = __builtin_amdgcn_wmma_f32_16x16x32_f16(
                    false, a, false, bfr, (short)0, acc[n], false, false);
            }
        }

        // C/D layout: element r of lane L -> M = r + 8*(L>>4), N = (L&15)
#pragma unroll
        for (int n = 0; n < 4; ++n) {
#pragma unroll
            for (int r = 0; r < 8; ++r) {
                int row = tile * 16 + r + 8 * khalf;
                int col = n * 16 + ncol;
                Z[(long)row * HID + col] = (_Float16)fmaxf(acc[n][r], 0.0f);
            }
        }
    }
}

// ---------------------------------------------------------------------
// Kernel 4: X[i, 0:64]  = Z[idx1[i]]
//           X[i, 64:128] = mean_j Z[idx2[i*K+j]]     (f32 accumulate)
// 64 threads per row i, one dim each.
// ---------------------------------------------------------------------
__global__ void build_x_kernel(const int* __restrict__ idx1,
                               const int* __restrict__ idx2,
                               const _Float16* __restrict__ Z,
                               _Float16* __restrict__ X) {
    int t = blockIdx.x * blockDim.x + threadIdx.x;
    int i = t >> 6;
    int d = t & 63;
    if (i >= BATCH * KNB) return;

    int n1 = idx1[i];
    X[(long)i * (2 * HID) + d] = Z[(long)n1 * HID + d];

    float s = 0.0f;
    const int* row = idx2 + i * KNB;
#pragma unroll
    for (int j = 0; j < KNB; ++j) {
        s += (float)Z[(long)row[j] * HID + d];
    }
    X[(long)i * (2 * HID) + HID + d] = (_Float16)(s * (1.0f / KNB));
}

// ---------------------------------------------------------------------
// Kernel 5: H1 = relu(X @ h1_w^T + h1_b)   [B*K x 64], f16 in/out.
// Same WMMA tiling as encode; A already f16 (two v8h chunk loads);
// weights staged in LDS.
// ---------------------------------------------------------------------
__global__ void h1_kernel(const _Float16* __restrict__ X,
                          const _Float16* __restrict__ h1w_h,
                          const float* __restrict__ h1_b,
                          _Float16* __restrict__ H1,
                          int n_tiles) {
    __shared__ _Float16 sB[HID * (2 * HID)];     // 16 KB
    fill_lds_weights(sB, h1w_h);
    __syncthreads();

    int wave = threadIdx.x >> 5;
    int lane = threadIdx.x & 31;
    int tile = blockIdx.x * (blockDim.x >> 5) + wave;
    if (tile < n_tiles) {
        int mrow  = tile * 16 + (lane & 15);
        int khalf = lane >> 4;
        int ncol  = lane & 15;

        v8f acc[4];
#pragma unroll
        for (int n = 0; n < 4; ++n) acc[n] = (v8f){};

#pragma unroll
        for (int kk = 0; kk < 4; ++kk) {
            int k0 = kk * 32 + khalf * 16;
            v8h lo = *(const v8h*)(X + (long)mrow * (2 * HID) + k0);
            v8h hi = *(const v8h*)(X + (long)mrow * (2 * HID) + k0 + 8);
            v16h a;
#pragma unroll
            for (int e = 0; e < 8; ++e) { a[e] = lo[e]; a[8 + e] = hi[e]; }

            int bk = kk * 32 + khalf * 16;
#pragma unroll
            for (int n = 0; n < 4; ++n) {
                v16h bfr = *(const v16h*)(sB + (n * 16 + ncol) * (2 * HID) + bk);
                acc[n] = __builtin_amdgcn_wmma_f32_16x16x32_f16(
                    false, a, false, bfr, (short)0, acc[n], false, false);
            }
        }

#pragma unroll
        for (int n = 0; n < 4; ++n) {
            float bias = h1_b[n * 16 + ncol];    // bias depends on column only
#pragma unroll
            for (int r = 0; r < 8; ++r) {
                int row = tile * 16 + r + 8 * khalf;
                int col = n * 16 + ncol;
                H1[(long)row * HID + col] = (_Float16)fmaxf(acc[n][r] + bias, 0.0f);
            }
        }
    }
}

// ---------------------------------------------------------------------
// Kernel 6: head. Per batch element b (one 64-thread block):
//   avg = mean_k H1[b,k,:];  h0 = relu(avg @ h2_w^T + h2_b);
//   out[b] = h0 @ out_w^T + out_b
// ---------------------------------------------------------------------
__global__ void head_kernel(const _Float16* __restrict__ H1,
                            const float* __restrict__ h2_w,
                            const float* __restrict__ h2_b,
                            const float* __restrict__ out_w,
                            const float* __restrict__ out_b,
                            float* __restrict__ out) {
    __shared__ float avg_s[HID];
    __shared__ float h0_s[HID];
    int b = blockIdx.x;
    int d = threadIdx.x;                          // 0..63

    float s = 0.0f;
#pragma unroll
    for (int k = 0; k < KNB; ++k) {
        s += (float)H1[((long)b * KNB + k) * HID + d];
    }
    avg_s[d] = s * (1.0f / KNB);
    __syncthreads();

    float acc = h2_b[d];
#pragma unroll 8
    for (int c = 0; c < HID; ++c) acc += avg_s[c] * h2_w[d * HID + c];
    h0_s[d] = fmaxf(acc, 0.0f);
    __syncthreads();

    if (d < OUTD) {
        float o = out_b[d];
#pragma unroll 8
        for (int c = 0; c < HID; ++c) o += h0_s[c] * out_w[d * HID + c];
        out[(long)b * OUTD + d] = o;
    }
}

// ---------------------------------------------------------------------
extern "C" void kernel_launch(void* const* d_in, const int* in_sizes, int n_in,
                              void* d_out, int out_size, void* d_ws, size_t ws_size,
                              hipStream_t stream) {
    const int*   idx0    = (const int*)  d_in[0];
    const int*   buf1    = (const int*)  d_in[1];
    const int*   buf2    = (const int*)  d_in[2];
    const int*   indptr  = (const int*)  d_in[3];
    const int*   indices = (const int*)  d_in[4];
    const int*   degrees = (const int*)  d_in[5];
    const float* emb     = (const float*)d_in[6];
    const float* enc_w   = (const float*)d_in[7];
    const float* h1_w    = (const float*)d_in[8];
    const float* h1_b    = (const float*)d_in[9];
    const float* h2_w    = (const float*)d_in[10];
    const float* h2_b    = (const float*)d_in[11];
    const float* out_w   = (const float*)d_in[12];
    const float* out_b   = (const float*)d_in[13];
    float* out = (float*)d_out;

    // ---- workspace layout (256B-aligned offsets) ----
    char* ws = (char*)d_ws;
    int*      idx1   = (int*)(ws + 0);            //   196,608 B
    int*      idx2   = (int*)(ws + 196608);       // 2,359,296 B
    _Float16* encw_h = (_Float16*)(ws + 2555904); //    16,384 B
    _Float16* h1w_h  = (_Float16*)(ws + 2572288); //    16,384 B
    _Float16* Z      = (_Float16*)(ws + 2588672); // 12,800,000 B
    _Float16* X      = (_Float16*)(ws + 15388672);// 12,582,912 B
    _Float16* H1     = (_Float16*)(ws + 27971584);//  6,291,456 B  (end ~34.3 MB)

    // 1) sampling: BATCH*KNB threads
    {
        int n = BATCH * KNB;
        sample_kernel<<<(n + 255) / 256, 256, 0, stream>>>(
            idx0, buf1, buf2, indptr, indices, degrees, idx1, idx2);
    }
    // 2) weight conversion
    {
        int n = 2 * HID * FEAT;
        cvt_weights_kernel<<<(n + 255) / 256, 256, 0, stream>>>(
            enc_w, h1_w, encw_h, h1w_h);
    }
    // 3) encoder GEMM: 6250 M-tiles, 8 waves (tiles) per 256-thread block
    {
        int n_tiles = N_NODES / 16;               // 6250 exactly
        int blocks  = (n_tiles + 7) / 8;
        encode_kernel<<<blocks, 256, 0, stream>>>(emb, encw_h, Z, n_tiles);
    }
    // 4) gather + 2-hop mean: 64 threads per row
    {
        long n = (long)BATCH * KNB * 64;
        build_x_kernel<<<(int)((n + 255) / 256), 256, 0, stream>>>(idx1, idx2, Z, X);
    }
    // 5) h1 GEMM: 3072 M-tiles
    {
        int n_tiles = (BATCH * KNB) / 16;         // 3072 exactly
        int blocks  = (n_tiles + 7) / 8;
        h1_kernel<<<blocks, 256, 0, stream>>>(X, h1w_h, h1_b, H1, n_tiles);
    }
    // 6) head: one 64-thread block per batch element
    {
        head_kernel<<<BATCH, 64, 0, stream>>>(H1, h2_w, h2_b, out_w, out_b, out);
    }
}
